// BlocksCore_44289702756726
// MI455X (gfx1250) — matrix operand order repaired and checked by reading, not compiled, not run on hardware
//
#include <hip/hip_runtime.h>
#include <cstdint>
#include <cstddef>

// ---------------------------------------------------------------------------
// Problem constants (match reference)
// ---------------------------------------------------------------------------
constexpr int BATCH   = 2048;
constexpr int NINP    = 1024;
constexpr int NHID    = 2048;
constexpr int NB      = 8;      // NB_OUT
constexpr int TOPK    = 4;
constexpr int BS      = 256;    // BS_OUT
constexpr int ATTOUT  = 1024;   // BS*4
constexpr int DK1     = 64;
constexpr int G3      = 768;    // 3*BS

// ---------------------------------------------------------------------------
// WMMA types (CDNA5 / gfx1250, wave32)
// ---------------------------------------------------------------------------
typedef __attribute__((ext_vector_type(16))) __bf16 v16bf;
typedef __attribute__((ext_vector_type(8)))  float  v8f;

union Frag { v16bf v; uint4 q[2]; };

// ---------------------------------------------------------------------------
// Generic batched GEMM:  C[g,m,n] = sum_k A[g,m,k] * B[g,k,n]
//   A fp32 row-major (lda); B fp32 KxN (TRANSB=0) or NxK (TRANSB=1);
//   C fp32 (ldc); per-group element strides sAg/sBg/sCg.
//   Tile 128x64x32, 256 threads = 8 waves, one 16x64 strip per wave.
//   fp32 -> bf16 at LDS staging; v_wmma_f32_16x16x32_bf16, f32 accumulate.
//   Double-buffered LDS pipeline: next tile's global loads are issued before
//   the current tile's ds_load/wmma stream; cvt+ds_store after the wmmas.
//   All M % 128 == 0, N % 64 == 0, K % 32 == 0 for every call here.
// ---------------------------------------------------------------------------
constexpr int BM = 128, BN = 64, BK = 32;
constexpr int NPA = BM * (BK / 2) / 256;   // 8 float2 loads per thread (A)
constexpr int NPB = BN * (BK / 2) / 256;   // 4 float2 loads per thread (B)

template<int TRANSB>
__global__ __launch_bounds__(256)
void gemm_bf16_wmma(const float* __restrict__ A, const float* __restrict__ Bm,
                    float* __restrict__ C, int K, int lda, int ldb, int ldc,
                    long long sAg, long long sBg, long long sCg)
{
    __shared__ __align__(16) __bf16 As[2][BM * BK];   // 2 x 8KB
    __shared__ __align__(16) __bf16 Bs[2][BN * BK];   // 2 x 4KB

    const int g = blockIdx.z;
    A  += (long long)g * sAg;
    Bm += (long long)g * sBg;
    C  += (long long)g * sCg;

    const int m0    = blockIdx.y * BM;
    const int n0    = blockIdx.x * BN;
    const int tid   = threadIdx.x;
    const int wave  = tid >> 5;      // 0..7  (wave32)
    const int lane  = tid & 31;
    const int row16 = lane & 15;
    const int half  = lane >> 4;

    // ---- per-thread staging bases (hoisted out of the k-loop) ----
    const float* pA[NPA]; int lA[NPA];
    #pragma unroll
    for (int p = 0; p < NPA; ++p) {
        int i  = tid + p * 256;
        int r  = i >> 4;              // tile row
        int kp = (i & 15) << 1;       // k pair
        pA[p] = A + (long long)(m0 + r) * lda + kp;
        lA[p] = r * BK + kp;
    }
    const float* pB[NPB]; int lB[NPB];
    #pragma unroll
    for (int p = 0; p < NPB; ++p) {
        int i  = tid + p * 256;
        int r  = i >> 4;
        int kp = (i & 15) << 1;
        if (TRANSB) pB[p] = Bm + (long long)(n0 + r) * ldb + kp;   // NxK: rows direct
        else        pB[p] = Bm + (long long)kp * ldb + n0 + r;     // KxN: transpose gather
        lB[p] = r * BK + kp;
    }

    float2 aReg[NPA], bReg[NPB];

    auto loadTile = [&](int k0) {
        #pragma unroll
        for (int p = 0; p < NPA; ++p)
            aReg[p] = *(const float2*)(pA[p] + k0);
        #pragma unroll
        for (int p = 0; p < NPB; ++p) {
            if (TRANSB) {
                bReg[p] = *(const float2*)(pB[p] + k0);
            } else {
                const float* qp = pB[p] + (long long)k0 * ldb;
                bReg[p].x = qp[0];
                bReg[p].y = qp[ldb];
            }
        }
    };
    auto storeTile = [&](int buf) {
        #pragma unroll
        for (int p = 0; p < NPA; ++p) {
            union { __bf16 h[2]; unsigned u; } pk;
            pk.h[0] = (__bf16)aReg[p].x; pk.h[1] = (__bf16)aReg[p].y;
            *(unsigned*)&As[buf][lA[p]] = pk.u;
        }
        #pragma unroll
        for (int p = 0; p < NPB; ++p) {
            union { __bf16 h[2]; unsigned u; } pk;
            pk.h[0] = (__bf16)bReg[p].x; pk.h[1] = (__bf16)bReg[p].y;
            *(unsigned*)&Bs[buf][lB[p]] = pk.u;
        }
    };

    v8f acc[4] = {};

    // ---- pipeline prologue ----
    loadTile(0);
    storeTile(0);
    __syncthreads();

    const int nIter = K / BK;
    for (int it = 0; it < nIter; ++it) {
        const int cur = it & 1;
        const bool hasNext = (it + 1) < nIter;
        if (hasNext) loadTile((it + 1) * BK);      // issue global loads early

        // ---- fragments from LDS, then wmma (batched 2 per dscnt wait) ----
        // A frag 16x32 bf16: per-lane-half contiguous K run of 16
        // (K = 16*half .. 16*half+15), matching B.
        const __bf16* as = As[cur];
        const __bf16* bs = Bs[cur];
        Frag af;
        const __bf16* arow = as + (wave * 16 + row16) * BK;
        af.q[0] = *(const uint4*)(arow + 16 * half);
        af.q[1] = *(const uint4*)(arow + 16 * half + 8);

        // B frag 32x16 bf16: lane = column n, per-lane-half contiguous K run
        // of 16 (K = 16*half .. 16*half+15).
        #pragma unroll
        for (int np = 0; np < 2; ++np) {
            Frag b0, b1;
            const __bf16* br0 = bs + ((np * 2) * 16 + row16) * BK + 16 * half;
            const __bf16* br1 = br0 + 16 * BK;
            b0.q[0] = *(const uint4*)(br0);
            b0.q[1] = *(const uint4*)(br0 + 8);
            b1.q[0] = *(const uint4*)(br1);
            b1.q[1] = *(const uint4*)(br1 + 8);
            acc[np * 2 + 0] = __builtin_amdgcn_wmma_f32_16x16x32_bf16(
                false, af.v, false, b0.v, (short)0, acc[np * 2 + 0], false, false);
            acc[np * 2 + 1] = __builtin_amdgcn_wmma_f32_16x16x32_bf16(
                false, af.v, false, b1.v, (short)0, acc[np * 2 + 1], false, false);
        }

        if (hasNext) storeTile(cur ^ 1);           // cvt+store after the wmmas
        __syncthreads();
    }

    // ---- epilogue: C/D layout — VGPR r holds M=r (lanes 0-15) / M=r+8 ----
    #pragma unroll
    for (int nt = 0; nt < 4; ++nt) {
        int n = n0 + nt * 16 + row16;
        #pragma unroll
        for (int r = 0; r < 8; ++r) {
            int m = m0 + wave * 16 + r + 8 * half;
            C[(long long)m * ldc + n] = acc[nt][r];
        }
    }
}

// ---------------------------------------------------------------------------
// Input attention scores + top-k block mask.
//   null-key logit is 0: softmax([0,l1]) => score0 = 1/(1+e^l1), a1 = 1-score0
//   drop the NB-TOPK largest score0 (ties: lower index dropped first).
// ---------------------------------------------------------------------------
__global__ __launch_bounds__(256)
void score_mask_kernel(const float* __restrict__ q1, const float* __restrict__ kk,
                       float* __restrict__ att1, float* __restrict__ mblk)
{
    __shared__ float sc[256];
    int t = threadIdx.x;
    int b = blockIdx.x * 32 + (t >> 3);
    int n = t & 7;
    const float* qr = q1 + (long long)b * (NB * DK1) + n * DK1;
    const float* kr = kk + (long long)b * DK1;
    float dot = 0.f;
    #pragma unroll 8
    for (int k = 0; k < DK1; ++k) dot += qr[k] * kr[k];
    float l1 = dot * 0.125f;                 // /sqrt(64)
    float e  = __expf(l1);
    float s0 = 1.f / (1.f + e);              // score (null attention)
    float a1 = 1.f - s0;                     // weight on real input
    sc[t] = s0;
    __syncthreads();
    const float* grp = &sc[t & ~7];
    int rank = 0;
    #pragma unroll
    for (int j = 0; j < 8; ++j) {
        float v = grp[j];
        rank += (v > s0) || (v == s0 && j < n);
    }
    float m = (rank < (NB - TOPK)) ? 0.f : 1.f;   // rank<4 => among 4 largest => dropped
    att1[b * 8 + n] = a1;
    mblk[b * 8 + n] = m;
}

// ---------------------------------------------------------------------------
// Block GRU elementwise:  gi = a * G + bih ; gh already GEMM'd ; gates.
// ---------------------------------------------------------------------------
__global__ __launch_bounds__(256)
void gru_kernel(const float* __restrict__ G, const float* __restrict__ gh,
                const float* __restrict__ att1, const float* __restrict__ hx,
                const float* __restrict__ bih, const float* __restrict__ bhh,
                float* __restrict__ hxnew)
{
    long long idx = (long long)blockIdx.x * 256 + threadIdx.x;   // < B*8*256
    int j = (int)(idx & 255);
    int n = (int)((idx >> 8) & 7);
    long long b = idx >> 11;
    float a = att1[b * 8 + n];
    const float* Gr = G  + b * (NB * G3) + n * G3 + j;
    const float* Hr = gh + b * (NB * G3) + n * G3 + j;
    const float* bi = bih + n * G3 + j;
    const float* bh = bhh + n * G3 + j;
    float i_r = a * Gr[0]        + bi[0];
    float i_z = a * Gr[BS]       + bi[BS];
    float i_n = a * Gr[2 * BS]   + bi[2 * BS];
    float h_r = Hr[0]      + bh[0];
    float h_z = Hr[BS]     + bh[BS];
    float h_n = Hr[2 * BS] + bh[2 * BS];
    float r  = 1.f / (1.f + __expf(-(i_r + h_r)));
    float z  = 1.f / (1.f + __expf(-(i_z + h_z)));
    float nn = tanhf(i_n + r * h_n);
    float hp = hx[b * NHID + n * BS + j];
    hxnew[idx] = (1.f - z) * nn + z * hp;
}

// ---------------------------------------------------------------------------
// Second-stage attention over the 8 blocks, 4 heads x 16 dims, per batch row.
// One lane per (q,head); 8 batch rows per 256-thread block.
// ---------------------------------------------------------------------------
__global__ __launch_bounds__(256)
void att2_kernel(const float* __restrict__ q2, const float* __restrict__ k2,
                 const float* __restrict__ v2, float* __restrict__ out2)
{
    int t    = threadIdx.x;
    long long b = (long long)blockIdx.x * 8 + (t >> 5);
    int lane = t & 31;
    int qi   = lane >> 2;    // block (query) 0..7
    int h    = lane & 3;     // head 0..3
    const float* qr = q2 + b * 512 + qi * 64 + h * 16;
    float qv[16];
    #pragma unroll
    for (int d = 0; d < 16; ++d) qv[d] = qr[d];
    float logit[8]; float mx = -1e30f;
    #pragma unroll
    for (int k = 0; k < 8; ++k) {
        const float* kr = k2 + b * 512 + k * 64 + h * 16;
        float dd = 0.f;
        #pragma unroll
        for (int d = 0; d < 16; ++d) dd += qv[d] * kr[d];
        dd *= 0.25f;                          // /sqrt(16)
        logit[k] = dd; mx = fmaxf(mx, dd);
    }
    float sum = 0.f;
    #pragma unroll
    for (int k = 0; k < 8; ++k) { logit[k] = __expf(logit[k] - mx); sum += logit[k]; }
    float inv = 1.f / sum;
    float o[16];
    #pragma unroll
    for (int d = 0; d < 16; ++d) o[d] = 0.f;
    #pragma unroll
    for (int k = 0; k < 8; ++k) {
        float p = logit[k] * inv;
        const float* vr = v2 + b * 512 + k * 64 + h * 16;
        #pragma unroll
        for (int d = 0; d < 16; ++d) o[d] += p * vr[d];
    }
    float* orow = out2 + b * 512 + qi * 64 + h * 16;
    #pragma unroll
    for (int d = 0; d < 16; ++d) orow[d] = o[d];
}

// ---------------------------------------------------------------------------
// Final gate / combine / mask; writes both outputs (hx_out, mask).
// ---------------------------------------------------------------------------
__global__ __launch_bounds__(256)
void final_kernel(const float* __restrict__ fcv, const float* __restrict__ gatev,
                  const float* __restrict__ fc_b, const float* __restrict__ gate_b,
                  const float* __restrict__ hxnew, const float* __restrict__ hx,
                  const float* __restrict__ mblk, float* __restrict__ out)
{
    long long idx = (long long)blockIdx.x * 256 + threadIdx.x;   // < B*NHID
    int j = (int)(idx & 255);
    long long bn = idx >> 8;                                     // b*8 + n
    float gv = gatev[bn * BS + j] + gate_b[j];
    float fv = fcv[bn * BS + j]   + fc_b[j];
    float gate = 1.f / (1.f + __expf(-gv));
    float hatt = gate * tanhf(fv);
    float hn = hxnew[idx] + hatt;
    float m  = mblk[bn];
    out[idx] = m * hn + (1.f - m) * hx[idx];
    out[(long long)BATCH * NHID + idx] = m;
}

// ---------------------------------------------------------------------------
// Orchestration
// ---------------------------------------------------------------------------
extern "C" void kernel_launch(void* const* d_in, const int* in_sizes, int n_in,
                              void* d_out, int out_size, void* d_ws, size_t ws_size,
                              hipStream_t stream)
{
    (void)in_sizes; (void)n_in; (void)out_size; (void)ws_size;

    const float* inp    = (const float*)d_in[0];
    const float* hx     = (const float*)d_in[1];
    const float* Wq1    = (const float*)d_in[2];
    const float* Wk1    = (const float*)d_in[3];
    const float* Wv1    = (const float*)d_in[4];
    const float* Wq2    = (const float*)d_in[5];
    const float* Wk2    = (const float*)d_in[6];
    const float* Wv2    = (const float*)d_in[7];
    const float* fc_w   = (const float*)d_in[8];
    const float* fc_b   = (const float*)d_in[9];
    const float* gate_w = (const float*)d_in[10];
    const float* gate_b = (const float*)d_in[11];
    const float* Wih    = (const float*)d_in[12];
    const float* Whh    = (const float*)d_in[13];
    const float* bih    = (const float*)d_in[14];
    const float* bhh    = (const float*)d_in[15];
    float* out = (float*)d_out;

    // ---- workspace layout (fp32), with overlays of dead buffers ----
    float* ws = (float*)d_ws;
    size_t o = 0;
    float* v     = ws + o; o += (size_t)BATCH * ATTOUT;        //  8 MB
    float* kk    = ws + o; o += (size_t)BATCH * DK1;           //  0.5 MB
    float* q1    = ws + o; o += (size_t)BATCH * NB * DK1;      //  4 MB
    float* att1  = ws + o; o += (size_t)BATCH * NB;
    float* mblk  = ws + o; o += (size_t)BATCH * NB;
    float* G     = ws + o; o += (size_t)BATCH * NB * G3;       // 48 MB
    float* gh    = ws + o; o += (size_t)BATCH * NB * G3;       // 48 MB
    float* hxnew = ws + o; o += (size_t)BATCH * NHID;          // 16 MB
    float* q2    = q1;                                         // overlay: q1 dead
    float* k2    = ws + o; o += (size_t)BATCH * NB * 64;       //  4 MB
    float* v2    = ws + o; o += (size_t)BATCH * NB * 64;       //  4 MB
    float* out2  = ws + o; o += (size_t)BATCH * NB * 64;       //  4 MB
    float* fcv   = G;                                          // overlay: G dead
    float* gatev = G + (size_t)BATCH * NB * BS;                // after GRU

    dim3 blk(256);

    // 1) kk = inp @ Wk1[1]                 (2048x1024 @ 1024x64)
    gemm_bf16_wmma<0><<<dim3(1, BATCH / BM, 1), blk, 0, stream>>>(
        inp, Wk1 + (size_t)NINP * DK1, kk,
        NINP, NINP, DK1, DK1, 0, 0, 0);

    // 2) v = inp @ Wv1[1]                  (2048x1024 @ 1024x1024)
    gemm_bf16_wmma<0><<<dim3(ATTOUT / BN, BATCH / BM, 1), blk, 0, stream>>>(
        inp, Wv1 + (size_t)NINP * ATTOUT, v,
        NINP, NINP, ATTOUT, ATTOUT, 0, 0, 0);

    // 3) q1[g] = hx_block[g] @ Wq1[g]      (8 x: 2048x256 @ 256x64)
    gemm_bf16_wmma<0><<<dim3(1, BATCH / BM, NB), blk, 0, stream>>>(
        hx, Wq1, q1,
        BS, NHID, DK1, NB * DK1, BS, (long long)BS * DK1, DK1);

    // 4) input-attention sigmoid + top-k block mask
    score_mask_kernel<<<dim3(BATCH / 32), blk, 0, stream>>>(q1, kk, att1, mblk);

    // 5) G[g] = v @ Wih[g]^T               (8 x: 2048x1024 @ 1024x768) -- dominant GEMM
    gemm_bf16_wmma<1><<<dim3(G3 / BN, BATCH / BM, NB), blk, 0, stream>>>(
        v, Wih, G,
        ATTOUT, ATTOUT, ATTOUT, NB * G3, 0, (long long)G3 * ATTOUT, G3);

    // 6) gh[g] = hx_block[g] @ Whh[g]^T    (8 x: 2048x256 @ 256x768)
    gemm_bf16_wmma<1><<<dim3(G3 / BN, BATCH / BM, NB), blk, 0, stream>>>(
        hx, Whh, gh,
        BS, NHID, BS, NB * G3, BS, (long long)G3 * BS, G3);

    // 7) GRU gates -> hx_new
    gru_kernel<<<dim3((BATCH * NB * BS) / 256), blk, 0, stream>>>(
        G, gh, att1, hx, bih, bhh, hxnew);

    // 8) q2/k2/v2 = hx_new_block @ W{q,k,v}2   (8 x: 2048x256 @ 256x64 each)
    gemm_bf16_wmma<0><<<dim3(1, BATCH / BM, NB), blk, 0, stream>>>(
        hxnew, Wq2, q2, BS, NHID, 64, NB * 64, BS, (long long)BS * 64, 64);
    gemm_bf16_wmma<0><<<dim3(1, BATCH / BM, NB), blk, 0, stream>>>(
        hxnew, Wk2, k2, BS, NHID, 64, NB * 64, BS, (long long)BS * 64, 64);
    gemm_bf16_wmma<0><<<dim3(1, BATCH / BM, NB), blk, 0, stream>>>(
        hxnew, Wv2, v2, BS, NHID, 64, NB * 64, BS, (long long)BS * 64, 64);

    // 9) block-to-block multi-head attention -> out2
    att2_kernel<<<dim3(BATCH / 8), blk, 0, stream>>>(q2, k2, v2, out2);

    // 10) fcv = out2_flat @ fc_w ; gatev = out2_flat @ gate_w  (16384x64 @ 64x256)
    gemm_bf16_wmma<0><<<dim3(BS / BN, (BATCH * NB) / BM, 1), blk, 0, stream>>>(
        out2, fc_w, fcv, 64, 64, BS, BS, 0, 0, 0);
    gemm_bf16_wmma<0><<<dim3(BS / BN, (BATCH * NB) / BM, 1), blk, 0, stream>>>(
        out2, gate_w, gatev, 64, 64, BS, BS, 0, 0, 0);

    // 11) gate/tanh combine + block mask, write hx_out and mask
    final_kernel<<<dim3((BATCH * NHID) / 256), blk, 0, stream>>>(
        fcv, gatev, fc_b, gate_b, hxnew, hx, mblk, out);
}
